// LGFIBlock_45973329936929
// MI455X (gfx1250) — matrix operand order repaired and checked by reading, not compiled, not run on hardware
//
#include <hip/hip_runtime.h>
#include <hip/hip_bf16.h>
#include <math.h>
#include <cstdint>

// ---------------------------------------------------------------------------
// LGFI block for MI455X (gfx1250): all matmuls via v_wmma_f32_16x16x32_f16,
// LDS staging via the Tensor Data Mover (tensor_load_to_lds + TENSORcnt).
// B=16, C=192, H=W=32 (N=1024 tokens), heads=4, d=48 (padded to 64 for WMMA).
// ---------------------------------------------------------------------------

typedef __attribute__((ext_vector_type(16))) _Float16 v16h;
typedef __attribute__((ext_vector_type(8)))  _Float16 v8h;
typedef __attribute__((ext_vector_type(8)))  float    v8f;
typedef unsigned int __attribute__((ext_vector_type(4))) u32x4;
typedef int          __attribute__((ext_vector_type(8))) i32x8;
typedef int          __attribute__((ext_vector_type(4))) i32x4;

__device__ __forceinline__ v16h cat8(v8h lo, v8h hi) {
  return __builtin_shufflevector(lo, hi, 0,1,2,3,4,5,6,7,8,9,10,11,12,13,14,15);
}
__device__ __forceinline__ v8f wmma16(v16h a, v16h b, v8f c) {
  // D = A(16x32 f16) * B(32x16 f16) + C(16x16 f32)
  return __builtin_amdgcn_wmma_f32_16x16x32_f16(false, a, false, b, (short)0, c, false, false);
}
__device__ __forceinline__ v16h ldb(const _Float16* p, int ld, int hb) {
  v16h v;
  #pragma unroll
  for (int s = 0; s < 16; ++s) v[s] = p[(hb * 8 + (s & 7) + (s >> 3) * 16) * ld];
  return v;
}

#define CC   192
#define NTOK 1024
#define NH   4
#define DH   48
#define DPAD 64

#if defined(__HIP_DEVICE_COMPILE__) && __has_builtin(__builtin_amdgcn_tensor_load_to_lds)
#define USE_TDM 1
#else
#define USE_TDM 0
#endif

#if USE_TDM
__device__ __forceinline__ unsigned lds_off(const void* p) {
  // generic LDS pointer: addr[31:0] is the LDS byte offset (ISA 10.2 aperture map)
  return (unsigned)(uintptr_t)p;
}

// 2D tile load (f16 elements): tile_d0 x tile_d1 from a row stride of stride0
// elements, landing contiguously (row-major) at LDS byte offset lds_addr.
__device__ __forceinline__ void tdm_load_2d(const _Float16* gaddr, unsigned lds_addr,
                                            unsigned tensor_d0, unsigned tensor_d1,
                                            unsigned tile_d0, unsigned tile_d1,
                                            unsigned stride0)
{
  unsigned long long ga = (unsigned long long)(uintptr_t)gaddr;
  u32x4 g0;
  g0[0] = 1u;                                            // count=1 (valid), user mode
  g0[1] = lds_addr;                                      // [63:32] lds_addr
  g0[2] = (unsigned)(ga & 0xFFFFFFFFull);                // [95:64] global_addr lo
  g0[3] = (unsigned)((ga >> 32) & 0x1FFFFFFull)          // [120:96] global_addr hi
        | (2u << 30);                                    // [127:126] type=2 (image)
  i32x8 g1;
  g1[0] = (int)(1u << 16);                               // data_size=1 -> 2 bytes
  g1[1] = (int)((tensor_d0 & 0xFFFFu) << 16);            // tensor_dim0[15:0] @48
  g1[2] = (int)((tensor_d0 >> 16) | ((tensor_d1 & 0xFFFFu) << 16));
  g1[3] = (int)((tensor_d1 >> 16) | (tile_d0 << 16));    // tile_dim0 @112
  g1[4] = (int)(tile_d1 & 0xFFFFu);                      // tile_dim1 @128, tile_dim2=0
  g1[5] = (int)stride0;                                  // tensor_dim0_stride[31:0]
  g1[6] = 0;                                             // stride0[47:32], stride1[15:0]
  g1[7] = 0;                                             // stride1[47:16]
  i32x4 z = {0, 0, 0, 0};                                // 2D: groups 2/3 disabled
#if defined(__clang_major__) && (__clang_major__ >= 23)
  i32x8 z8 = {0, 0, 0, 0, 0, 0, 0, 0};
  __builtin_amdgcn_tensor_load_to_lds(g0, g1, z, z, z8, 0);
#else
  __builtin_amdgcn_tensor_load_to_lds(g0, g1, z, z, 0);
#endif
}
#endif  // USE_TDM

// ---------------------------------------------------------------------------
// Kernel 0: f32 -> f16 conversion (x once, each weight matrix once).
// ---------------------------------------------------------------------------
__global__ __launch_bounds__(256) void cvt_f16_kernel(const float* __restrict__ src,
                                                      _Float16* __restrict__ dst, int n)
{
  int i = blockIdx.x * 256 + threadIdx.x;
  if (i < n) dst[i] = (_Float16)src[i];
}

// ---------------------------------------------------------------------------
// Shared staging: W (192x192 f16) + X strip (192 x 64 tokens f16) into LDS.
// TDM path: wave 0 issues two DMAs, waits TENSORcnt, block barrier.
// ---------------------------------------------------------------------------
__device__ __forceinline__ void stage_wx(const _Float16* __restrict__ wh,
                                         const _Float16* __restrict__ xh,
                                         int b, int tok0,
                                         _Float16* Wl, _Float16* Xl)
{
  const _Float16* xb = xh + ((size_t)b * CC) * NTOK + tok0;
#if USE_TDM
  if (threadIdx.x < 32) {
    tdm_load_2d(wh, lds_off(Wl), CC * CC, 1, CC * CC, 1, CC * CC);  // 1 row, 36864 elems
    tdm_load_2d(xb, lds_off(Xl), NTOK, CC, 64, CC, NTOK);           // 192 rows of 64
    __builtin_amdgcn_s_wait_tensorcnt(0);
  }
#else
  const int tid = threadIdx.x;
  for (int i = tid; i < CC * CC; i += 256) Wl[i] = wh[i];
  for (int i = tid; i < CC * 64; i += 256) Xl[i] = xb[(i >> 6) * NTOK + (i & 63)];
#endif
  __syncthreads();
}

// ---------------------------------------------------------------------------
// Kernel 1: 1x1 conv (GEMM [192x192]x[192x1024]) + BN + exact GELU -> d_out
// grid = B * 16 blocks, 256 threads (8 waves); block owns 64 tokens.
// ---------------------------------------------------------------------------
__global__ __launch_bounds__(256) void conv_bn_gelu_kernel(
    const _Float16* __restrict__ xh, const _Float16* __restrict__ wh,
    const float* __restrict__ bn_gamma, const float* __restrict__ bn_beta,
    const float* __restrict__ bn_mean,  const float* __restrict__ bn_var,
    float* __restrict__ out)
{
  __shared__ _Float16 Wl[CC * CC];   // 72 KB, row-major [o][c]
  __shared__ _Float16 Xl[CC * 64];   // 24 KB, row-major [c][tok]

  const int b    = blockIdx.x >> 4;
  const int tok0 = (blockIdx.x & 15) * 64;
  stage_wx(wh, xh, b, tok0, Wl, Xl);

  const int wave = threadIdx.x >> 5, lane = threadIdx.x & 31;
  const int lh = lane & 15, hb = lane >> 4;   // half-lane id / half select

  for (int j = 0; j < 6; ++j) {
    const int job = wave * 6 + j;             // 48 tiles: 12 M-tiles x 4 N-tiles
    const int mt = job >> 2, nt = job & 3;
    v8f acc = {};
    #pragma unroll
    for (int kc = 0; kc < 6; ++kc) {
      // A frag: row o = mt*16+lh ; K chunks {base, base+16}, base = kc*32+hb*8
      const _Float16* ar = &Wl[(mt * 16 + lh) * CC + kc * 32 + hb * 8];
      v16h a = cat8(*(const v8h*)ar, *(const v8h*)(ar + 16));
      // B frag: token column nt*16+lh ; K rows {base, base+16}, base = kc*32+hb*8
      const _Float16* br = &Xl[(kc * 32) * 64 + nt * 16 + lh];
      acc = wmma16(a, ldb(br, 64, hb), acc);
    }
    const int ncol = tok0 + nt * 16 + lh;
    #pragma unroll
    for (int r = 0; r < 8; ++r) {
      const int o = mt * 16 + r + hb * 8;     // C/D: lane<16 -> M=r, else M=r+8
      const float sc = bn_gamma[o] * rsqrtf(bn_var[o] + 1e-5f);
      const float sh = bn_beta[o] - bn_mean[o] * sc;
      const float v  = acc[r] * sc + sh;
      out[((size_t)b * CC + o) * NTOK + ncol] =
          0.5f * v * (1.0f + erff(v * 0.70710678118654752f));
    }
  }
}

// ---------------------------------------------------------------------------
// Kernel 2: QKV projection GEMM, f16 outputs into workspace.
// mode 0: Q -> [bh][tok][DPAD] (scaled by 1/sqrt(d))
// mode 1: K -> [bh][DPAD][tok] (transposed, ready as B-frag rows)
// mode 2: V -> [bh][tok][DPAD]
// ---------------------------------------------------------------------------
__global__ __launch_bounds__(256) void qkv_kernel(
    const _Float16* __restrict__ xh, const _Float16* __restrict__ wh,
    _Float16* __restrict__ dst, int mode, float scale)
{
  __shared__ _Float16 Wl[CC * CC];
  __shared__ _Float16 Xl[CC * 64];

  const int b    = blockIdx.x >> 4;
  const int tok0 = (blockIdx.x & 15) * 64;
  stage_wx(wh, xh, b, tok0, Wl, Xl);

  const int wave = threadIdx.x >> 5, lane = threadIdx.x & 31;
  const int lh = lane & 15, hb = lane >> 4;

  for (int j = 0; j < 6; ++j) {
    const int job = wave * 6 + j;
    const int mt = job >> 2, nt = job & 3;
    v8f acc = {};
    #pragma unroll
    for (int kc = 0; kc < 6; ++kc) {
      const _Float16* ar = &Wl[(mt * 16 + lh) * CC + kc * 32 + hb * 8];
      v16h a = cat8(*(const v8h*)ar, *(const v8h*)(ar + 16));
      const _Float16* br = &Xl[(kc * 32) * 64 + nt * 16 + lh];
      acc = wmma16(a, ldb(br, 64, hb), acc);
    }
    const int tok = tok0 + nt * 16 + lh;
    #pragma unroll
    for (int r = 0; r < 8; ++r) {
      const int o    = mt * 16 + r + hb * 8;
      const int head = o / DH, cl = o % DH;
      const int bh   = b * NH + head;
      const _Float16 v = (_Float16)(acc[r] * scale);
      if (mode == 1)
        dst[((size_t)bh * DPAD + cl) * NTOK + tok] = v;      // K^T layout
      else
        dst[((size_t)bh * NTOK + tok) * DPAD + cl] = v;      // Q / V layout
    }
  }
}

// ---------------------------------------------------------------------------
// Kernel 3: flash attention, one wave per (b, head, 16-query tile).
// grid = 1024 blocks x 128 threads (4 waves). 32 chunks of 32 keys.
// ---------------------------------------------------------------------------
__global__ __launch_bounds__(128) void attn_kernel(
    const _Float16* __restrict__ qws, const _Float16* __restrict__ kws,
    const _Float16* __restrict__ vws, float* __restrict__ gws)
{
  __shared__ _Float16 Pl[4][16 * 32];   // per-wave P bounce (C/D -> A relayout)

  const int wave = threadIdx.x >> 5, lane = threadIdx.x & 31;
  const int gw   = blockIdx.x * 4 + wave;   // 0..4095
  const int qt   = gw & 63;
  const int bh   = gw >> 6;                 // b*4+head
  const int b    = bh >> 2, head = bh & 3;
  const int q0   = qt * 16;
  const int lh   = lane & 15, hb = lane >> 4;

  const _Float16* Q  = qws + (size_t)bh * NTOK * DPAD;
  const _Float16* Kt = kws + (size_t)bh * DPAD * NTOK;
  const _Float16* V  = vws + (size_t)bh * NTOK * DPAD;

  // Q A-frags (d = 0..31 and 32..63), already scaled by 1/sqrt(d)
  const _Float16* qr = Q + (size_t)(q0 + lh) * DPAD;
  v16h aq0 = cat8(*(const v8h*)(qr + hb * 8),      *(const v8h*)(qr + hb * 8 + 16));
  v16h aq1 = cat8(*(const v8h*)(qr + 32 + hb * 8), *(const v8h*)(qr + 32 + hb * 8 + 16));

  v8f o0 = {}, o1 = {}, o2 = {};
  float m[8], l[8];
  #pragma unroll
  for (int r = 0; r < 8; ++r) { m[r] = -1e30f; l[r] = 0.0f; }
  _Float16* P = Pl[wave];

  for (int kc = 0; kc < 32; ++kc) {
    const int k0 = kc * 32;
    // ---- scores: two 16-key tiles, K-dim = DPAD in two 32-chunks ----
    v8f s0 = {}, s1 = {};
    s0 = wmma16(aq0, ldb(Kt + k0 + lh, NTOK, hb), s0);
    s0 = wmma16(aq1, ldb(Kt + (size_t)32 * NTOK + k0 + lh, NTOK, hb), s0);
    s1 = wmma16(aq0, ldb(Kt + k0 + 16 + lh, NTOK, hb), s1);
    s1 = wmma16(aq1, ldb(Kt + (size_t)32 * NTOK + k0 + 16 + lh, NTOK, hb), s1);

    // ---- online softmax (row stats map: reg r <-> rows r / r+8) ----
    #pragma unroll
    for (int r = 0; r < 8; ++r) {
      float cm = fmaxf(s0[r], s1[r]);
      cm = fmaxf(cm, __shfl_xor(cm, 1, 32));
      cm = fmaxf(cm, __shfl_xor(cm, 2, 32));
      cm = fmaxf(cm, __shfl_xor(cm, 4, 32));
      cm = fmaxf(cm, __shfl_xor(cm, 8, 32));
      const float mn = fmaxf(m[r], cm);
      const float al = __expf(m[r] - mn);
      m[r] = mn;
      const float p0 = __expf(s0[r] - mn);
      const float p1 = __expf(s1[r] - mn);
      s0[r] = p0; s1[r] = p1;
      float sum = p0 + p1;
      sum += __shfl_xor(sum, 1, 32);
      sum += __shfl_xor(sum, 2, 32);
      sum += __shfl_xor(sum, 4, 32);
      sum += __shfl_xor(sum, 8, 32);
      l[r] = l[r] * al + sum;
      o0[r] *= al; o1[r] *= al; o2[r] *= al;
    }

    // ---- P: C/D layout -> A layout via LDS ----
    __syncthreads();
    #pragma unroll
    for (int r = 0; r < 8; ++r) {
      P[(r + hb * 8) * 32 + lh]      = (_Float16)s0[r];
      P[(r + hb * 8) * 32 + 16 + lh] = (_Float16)s1[r];
    }
    __syncthreads();
    const _Float16* pr = P + lh * 32 + hb * 8;
    v16h pa = cat8(*(const v8h*)pr, *(const v8h*)(pr + 16));

    // ---- O += P * V  (skip d-tile 3: padded zeros) ----
    const _Float16* vr = V + (size_t)k0 * DPAD + lh;
    o0 = wmma16(pa, ldb(vr, DPAD, hb),      o0);
    o1 = wmma16(pa, ldb(vr + 16, DPAD, hb), o1);
    o2 = wmma16(pa, ldb(vr + 32, DPAD, hb), o2);
  }

  // ---- normalize and scatter to g[b][tok][c] ----
  #pragma unroll
  for (int r = 0; r < 8; ++r) {
    const float inv = 1.0f / l[r];
    const int tok = q0 + r + hb * 8;
    float* gr = gws + ((size_t)b * NTOK + tok) * CC + head * DH;
    gr[0 * 16 + lh] = o0[r] * inv;
    gr[1 * 16 + lh] = o1[r] * inv;
    gr[2 * 16 + lh] = o2[r] * inv;
  }
}

// ---------------------------------------------------------------------------
// Kernel 4: LayerNorm over C=192 and add into d_out (which holds local path).
// One wave per (b, tok): 16384 rows, 8 waves/block -> 2048 blocks.
// ---------------------------------------------------------------------------
__global__ __launch_bounds__(256) void ln_add_kernel(
    const float* __restrict__ gws, const float* __restrict__ ln_gamma,
    const float* __restrict__ ln_beta, float* __restrict__ out)
{
  const int wave = threadIdx.x >> 5, lane = threadIdx.x & 31;
  const int row  = blockIdx.x * 8 + wave;   // b*1024 + tok
  const int b    = row >> 10, tok = row & 1023;
  const float* g = gws + (size_t)row * CC;

  float v[6], sum = 0.0f;
  #pragma unroll
  for (int i = 0; i < 6; ++i) { v[i] = g[i * 32 + lane]; sum += v[i]; }
  sum += __shfl_xor(sum, 1, 32);
  sum += __shfl_xor(sum, 2, 32);
  sum += __shfl_xor(sum, 4, 32);
  sum += __shfl_xor(sum, 8, 32);
  sum += __shfl_xor(sum, 16, 32);
  const float mu = sum * (1.0f / CC);

  float var = 0.0f;
  #pragma unroll
  for (int i = 0; i < 6; ++i) { const float d = v[i] - mu; var += d * d; }
  var += __shfl_xor(var, 1, 32);
  var += __shfl_xor(var, 2, 32);
  var += __shfl_xor(var, 4, 32);
  var += __shfl_xor(var, 8, 32);
  var += __shfl_xor(var, 16, 32);
  const float rstd = rsqrtf(var * (1.0f / CC) + 1e-5f);

  #pragma unroll
  for (int i = 0; i < 6; ++i) {
    const int c = i * 32 + lane;
    const float val = (v[i] - mu) * rstd * ln_gamma[c] + ln_beta[c];
    const size_t oi = ((size_t)b * CC + c) * NTOK + tok;
    out[oi] += val;
  }
}

// ---------------------------------------------------------------------------
extern "C" void kernel_launch(void* const* d_in, const int* in_sizes, int n_in,
                              void* d_out, int out_size, void* d_ws, size_t ws_size,
                              hipStream_t stream)
{
  (void)in_sizes; (void)n_in; (void)out_size; (void)ws_size;
  const float* x        = (const float*)d_in[0];
  const float* conv_w   = (const float*)d_in[1];
  const float* bn_gamma = (const float*)d_in[2];
  const float* bn_beta  = (const float*)d_in[3];
  const float* bn_mean  = (const float*)d_in[4];
  const float* bn_var   = (const float*)d_in[5];
  const float* wq       = (const float*)d_in[6];
  const float* wk       = (const float*)d_in[7];
  const float* wv       = (const float*)d_in[8];
  const float* ln_gamma = (const float*)d_in[9];
  const float* ln_beta  = (const float*)d_in[10];
  float* out = (float*)d_out;

  // workspace partition (all offsets MiB-aligned):
  //   0..8M   : Q f16   [bh][tok][64]
  //   8..16M  : K f16   [bh][64][tok]
  //   16..24M : V f16   [bh][tok][64]
  //   24..36M : g f32   [b][tok][192]
  //   36..42M : x  f16  [b][c][tok]
  //   42M..   : weights f16: conv_w | wq | wk | wv (72 KB each)
  char* ws = (char*)d_ws;
  _Float16* qws = (_Float16*)(ws);
  _Float16* kws = (_Float16*)(ws + (size_t)(8  << 20));
  _Float16* vws = (_Float16*)(ws + (size_t)(16 << 20));
  float*    gws = (float*)   (ws + (size_t)(24 << 20));
  _Float16* xh  = (_Float16*)(ws + (size_t)(36 << 20));
  _Float16* whc = (_Float16*)(ws + (size_t)(42 << 20));
  _Float16* whq = whc + CC * CC;
  _Float16* whk = whq + CC * CC;
  _Float16* whv = whk + CC * CC;

  // zero Q/K/V so the d=48->64 padding contributes nothing (capturable)
  hipMemsetAsync(ws, 0, (size_t)(24 << 20), stream);

  // one-time f32 -> f16 precision pass
  const int nx = 16 * CC * NTOK;          // 3,145,728
  const int nw = CC * CC;                 // 36,864
  cvt_f16_kernel<<<(nx + 255) / 256, 256, 0, stream>>>(x, xh, nx);
  cvt_f16_kernel<<<(nw + 255) / 256, 256, 0, stream>>>(conv_w, whc, nw);
  cvt_f16_kernel<<<(nw + 255) / 256, 256, 0, stream>>>(wq, whq, nw);
  cvt_f16_kernel<<<(nw + 255) / 256, 256, 0, stream>>>(wk, whk, nw);
  cvt_f16_kernel<<<(nw + 255) / 256, 256, 0, stream>>>(wv, whv, nw);

  conv_bn_gelu_kernel<<<256, 256, 0, stream>>>(xh, whc, bn_gamma, bn_beta,
                                               bn_mean, bn_var, out);
  const float inv_sqrt_d = 0.14433756729740646f;  // 1/sqrt(48)
  qkv_kernel<<<256, 256, 0, stream>>>(xh, whq, qws, 0, inv_sqrt_d);
  qkv_kernel<<<256, 256, 0, stream>>>(xh, whk, kws, 1, 1.0f);
  qkv_kernel<<<256, 256, 0, stream>>>(xh, whv, vws, 2, 1.0f);
  attn_kernel<<<1024, 128, 0, stream>>>(qws, kws, vws, gws);
  ln_add_kernel<<<2048, 256, 0, stream>>>(gws, ln_gamma, ln_beta, out);
}
